// NNUE_28192165331581
// MI455X (gfx1250) — compile-verified, matrix operand order checked by reading, not run on hardware
//
#include <hip/hip_runtime.h>

typedef __attribute__((ext_vector_type(16))) __bf16 v16bf;
typedef __attribute__((ext_vector_type(8)))  __bf16 v8bf;
typedef __attribute__((ext_vector_type(4)))  __bf16 v4bf;
typedef __attribute__((ext_vector_type(8)))  float  v8f;

#define NF     20480
#define NB     4096
#define NACC   128
#define NTILES 9                 // 8 accumulator tiles + 1 padded psqt tile
#define KSTEPS (NF / 32)         // 640
#define CHUNK  256               // K elements staged to LDS per iteration
#define NCHUNK (NF / CHUNK)      // 80
#define PITCH  (CHUNK + 8)       // bf16 halves per LDS row (bank-conflict padding)

static __device__ __forceinline__ __bf16 f2bf(float f) {
  union { float f; unsigned u; } c; c.f = f;
  unsigned short h = (unsigned short)((c.u + 0x8000u) >> 16);  // round-ish to nearest
  __bf16 r; __builtin_memcpy(&r, &h, 2);
  return r;
}

// ---------------------------------------------------------------------------
// Kernel 0: repack f32 weights into bf16, in the exact per-lane WMMA B layout.
// Entry index = (tile*KSTEPS + kstep)*32 + lane ; each entry = 16 bf16 (32 B).
// lane<16 : neuron = tile*16+lane, K = kstep*32 + {0..7, 16..23}
// lane>=16: neuron = tile*16+lane-16, K = kstep*32 + {8..15, 24..31}
// tile 8 = w_psqt padded: neurons 0..1 real, 2..15 zero.
// ---------------------------------------------------------------------------
__global__ void nnue_repack(const float* __restrict__ w_acc,
                            const float* __restrict__ w_psqt,
                            __bf16* __restrict__ pack) {
  int gid = blockIdx.x * blockDim.x + threadIdx.x;
  if (gid >= NTILES * KSTEPS * 32) return;
  int lane = gid & 31;
  int s    = (gid >> 5) % KSTEPS;
  int tile = gid / (KSTEPS * 32);
  int n    = lane & 15;
  int kbase = s * 32 + ((lane >> 4) << 3);

  bool valid;
  const float* src;
  if (tile < 8) { valid = true;     src = w_acc  + (size_t)(tile * 16 + n) * NF; }
  else          { valid = (n < 2);  src = w_psqt + (size_t)(valid ? n : 0) * NF; }

  __bf16* dst = pack + (size_t)gid * 16;
#pragma unroll
  for (int j = 0; j < 8; ++j)
    dst[j]     = valid ? f2bf(src[kbase + j])      : f2bf(0.f);
#pragma unroll
  for (int j = 0; j < 8; ++j)
    dst[8 + j] = valid ? f2bf(src[kbase + 16 + j]) : f2bf(0.f);
}

// ---------------------------------------------------------------------------
// Main kernel: one block = 16 batch rows; 9 waves = 8 w_acc N-tiles + psqt.
// Activations staged f32->bf16 through double-buffered LDS, shared by waves.
// ---------------------------------------------------------------------------
__global__ void __launch_bounds__(288) nnue_main(
    const float* __restrict__ white, const float* __restrict__ black,
    const __bf16* __restrict__ pack, const float* __restrict__ b_acc,
    const float* __restrict__ w_layer, float* __restrict__ out) {
  __shared__ __bf16 smW[2][16 * PITCH];
  __shared__ __bf16 smB[2][16 * PITCH];
  __shared__ float  sOut[32];

  const int t    = threadIdx.x;
  const int wave = t >> 5;
  const int lane = t & 31;
  const int mtile = blockIdx.x;
  const size_t rowbase = (size_t)mtile * 16 * NF;

  if (t < 32) sOut[t] = 0.f;

  v8f cw = {}; v8f cb = {};
  const v16bf* wptr = ((const v16bf*)pack) + (size_t)wave * KSTEPS * 32 + lane;

  auto stage = [&](int buf, int kc) {
    for (int i = t; i < 16 * (CHUNK / 4); i += 288) {
      int row = i >> 6;            // CHUNK/4 = 64 float4 per row
      int k   = (i & 63) << 2;
      const float4 fw = *(const float4*)(white + rowbase + (size_t)row * NF + kc + k);
      const float4 fb = *(const float4*)(black + rowbase + (size_t)row * NF + kc + k);
      v4bf bw = { f2bf(fw.x), f2bf(fw.y), f2bf(fw.z), f2bf(fw.w) };
      v4bf bb = { f2bf(fb.x), f2bf(fb.y), f2bf(fb.z), f2bf(fb.w) };
      *(v4bf*)&smW[buf][row * PITCH + k] = bw;
      *(v4bf*)&smB[buf][row * PITCH + k] = bb;
    }
  };

  union V16 { v16bf v; struct { v8bf lo, hi; } s; };

  auto compute = [&](int buf) {
    const __bf16* aW = &smW[buf][(lane & 15) * PITCH + ((lane >> 4) << 3)];
    const __bf16* aB = &smB[buf][(lane & 15) * PITCH + ((lane >> 4) << 3)];
#pragma unroll
    for (int s = 0; s < CHUNK / 32; ++s) {
      const int off = s * 32;
      V16 aw, ab;
      aw.s.lo = *(const v8bf*)(aW + off);        // K +0..7   (or +8..15)
      aw.s.hi = *(const v8bf*)(aW + off + 16);   // K +16..23 (or +24..31)
      ab.s.lo = *(const v8bf*)(aB + off);
      ab.s.hi = *(const v8bf*)(aB + off + 16);
      v16bf bt = *wptr; wptr += 32;              // next kstep, same lane
      cw = __builtin_amdgcn_wmma_f32_16x16x32_bf16(false, aw.v, false, bt,
                                                   (short)0, cw, false, false);
      cb = __builtin_amdgcn_wmma_f32_16x16x32_bf16(false, ab.v, false, bt,
                                                   (short)0, cb, false, false);
    }
  };

  stage(0, 0);
  __syncthreads();
  for (int c = 0; c < NCHUNK; ++c) {
    int buf = c & 1;
    if (c + 1 < NCHUNK) stage(buf ^ 1, (c + 1) * CHUNK);
    compute(buf);
    __syncthreads();
  }

  // Epilogue: bias + CReLU (acc tiles), project onto w_layer; psqt tile uses
  // identity weights with no bias/clamp so both paths unify.
  const int n = lane & 15;
  const bool isacc = (wave < 8);
  float bias = 0.f, wl0, wl1;
  if (isacc) {
    int neuron = wave * 16 + n;
    bias = b_acc[neuron];
    wl0 = w_layer[neuron];
    wl1 = w_layer[NACC + neuron];
  } else {
    wl0 = (n == 0) ? 1.f : 0.f;
    wl1 = (n == 1) ? 1.f : 0.f;
  }
#pragma unroll
  for (int r = 0; r < 8; ++r) {
    float aw = cw[r] + bias;
    float ab = cb[r] + bias;
    if (isacc) {
      aw = fminf(fmaxf(aw, 0.f), 1.f);
      ab = fminf(fmaxf(ab, 0.f), 1.f);
    }
    float d  = aw - ab;
    float p0 = d * wl0;
    float p1 = d * wl1;
#pragma unroll
    for (int m = 1; m < 16; m <<= 1) {           // reduce over 16-lane N halves
      p0 += __shfl_xor(p0, m, 32);
      p1 += __shfl_xor(p1, m, 32);
    }
    if (n == 0) {
      int mrow = r + ((lane >> 4) << 3);         // C layout: M = r (+8 hi half)
      atomicAdd(&sOut[mrow * 2 + 0], p0);
      atomicAdd(&sOut[mrow * 2 + 1], p1);
    }
  }
  __syncthreads();
  if (t < 32) out[(size_t)mtile * 32 + t] = sOut[t];
}

extern "C" void kernel_launch(void* const* d_in, const int* in_sizes, int n_in,
                              void* d_out, int out_size, void* d_ws, size_t ws_size,
                              hipStream_t stream) {
  const float* white   = (const float*)d_in[0];
  const float* black   = (const float*)d_in[1];
  const float* w_psqt  = (const float*)d_in[2];
  const float* w_acc   = (const float*)d_in[3];
  const float* b_acc   = (const float*)d_in[4];
  const float* w_layer = (const float*)d_in[5];
  float* out = (float*)d_out;
  __bf16* pack = (__bf16*)d_ws;   // needs NTILES*KSTEPS*32*16*2 B ~= 5.6 MB

  const int total = NTILES * KSTEPS * 32;
  nnue_repack<<<(total + 255) / 256, 256, 0, stream>>>(w_acc, w_psqt, pack);
  nnue_main<<<NB / 16, 288, 0, stream>>>(white, black, pack, b_acc, w_layer, out);
}
